// GNO_64226940944523
// MI455X (gfx1250) — compile-verified, matrix operand order checked by reading, not run on hardware
//
#include <hip/hip_runtime.h>
#include <hip/hip_bf16.h>

typedef __attribute__((ext_vector_type(16))) _Float16 v16h;
typedef __attribute__((ext_vector_type(8)))  _Float16 v8h;
typedef __attribute__((ext_vector_type(4)))  _Float16 v4h;
typedef __attribute__((ext_vector_type(8)))  float    v8f;

#define FIN  8
#define FOUT 8
#define HDIM 64
#define TASKS_PER_WAVE 4

// Hardware tanh: gfx1250 V_TANH_F32 (TRANS32, co-executes with WMMA).
__device__ __forceinline__ float fast_tanh(float x) {
#if __has_builtin(__builtin_amdgcn_tanhf)
    return __builtin_amdgcn_tanhf(x);
#elif __has_builtin(__builtin_amdgcn_tanh_f32)
    return __builtin_amdgcn_tanh_f32(x);
#else
    float r;
    asm volatile("v_tanh_f32 %0, %1\n\tv_nop" : "=v"(r) : "v"(x));
    return r;
#endif
}

// DPP8 xor-permute add: butterfly reduction within groups of 8 lanes.
constexpr int dpp8_sel(int a, int b, int c, int d, int e, int f, int g, int h) {
    return a | (b << 3) | (c << 6) | (d << 9) | (e << 12) | (f << 15) |
           (g << 18) | (h << 21);
}
#define DPP8_XOR1 dpp8_sel(1, 0, 3, 2, 5, 4, 7, 6)
#define DPP8_XOR2 dpp8_sel(2, 3, 0, 1, 6, 7, 4, 5)
#define DPP8_XOR4 dpp8_sel(4, 5, 6, 7, 0, 1, 2, 3)

template <int SEL>
__device__ __forceinline__ float dpp8_xadd(float x) {
#if __has_builtin(__builtin_amdgcn_mov_dpp8)
    int yi = __builtin_amdgcn_mov_dpp8(__builtin_bit_cast(int, x), SEL);
    return x + __builtin_bit_cast(float, yi);
#else
    return x + __shfl_xor(x, (SEL & 7), 32);
#endif
}

__device__ __forceinline__ float red8(float x) {
    x = dpp8_xadd<DPP8_XOR1>(x);
    x = dpp8_xadd<DPP8_XOR2>(x);
    x = dpp8_xadd<DPP8_XOR4>(x);
    return x;
}

// Load a 16-lane-striped 16x32 f16 fragment (A or B operand) from a row-major
// LDS tile. CDNA5 16-bit operand layout: lanes 0-15 hold K in [0,8)+[16,24),
// lanes 16-31 hold K in [8,16)+[24,32) (offset by kbase for K-stepping).
__device__ __forceinline__ v16h ldsFrag(const _Float16* row, int kbase, int khalf) {
    v16h a;
    *((v8h*)&a)       = *((const v8h*)(row + kbase + khalf));
    *(((v8h*)&a) + 1) = *((const v8h*)(row + kbase + 16 + khalf));
    return a;
}

__device__ __forceinline__ v8f wmma_f16(v16h a, v16h b, v8f c) {
    return __builtin_amdgcn_wmma_f32_16x16x32_f16(false, a, false, b,
                                                  (short)0, c, false, false);
}

__global__ void zero_kernel(float* __restrict__ p, long long n) {
    long long i = (long long)blockIdx.x * blockDim.x + threadIdx.x;
    if (i < n) p[i] = 0.0f;
}

// One wave32 processes 16 edges of one batch per task: 20->64->64->64 MLP on
// WMMA, then msg = mats(8x8) @ by with DPP8 reduction, scatter-add ksum/deg.
// b1 is folded into W1 via a constant-1 feature column (K=20), so the layer-1
// WMMA uses the inline-zero C operand. Grid-stride loop amortizes staging.
__global__ void __launch_bounds__(256)
msg_kernel(const float* __restrict__ batch, const float* __restrict__ points,
           const int* __restrict__ neigh,
           const float* __restrict__ W1, const float* __restrict__ b1,
           const float* __restrict__ W2, const float* __restrict__ b2,
           const float* __restrict__ W3, const float* __restrict__ b3,
           float* __restrict__ ksum, float* __restrict__ deg,
           int B, int N, int E, int tilesPerB)
{
    __shared__ alignas(16) _Float16 w1t[64 * 32];   // [W1; b1]^T, K pad 21->32
    __shared__ alignas(16) _Float16 w2t[64 * 64];   // W2^T
    __shared__ alignas(16) _Float16 w3t[64 * 64];   // W3^T
    __shared__ float sb2[64], sb3[64];
    __shared__ alignas(16) _Float16 featT[8][16 * 32];  // per-wave feature tile
    __shared__ alignas(16) _Float16 hT[8][16 * 64];     // per-wave activation tile
    __shared__ int dstT[8][16];

    const int tid = threadIdx.x;

    // ---- stage weights (transposed to [n][k], f16) once per block ----
    for (int idx = tid; idx < 64 * 32; idx += 256) {
        int n = idx >> 5, k = idx & 31;
        _Float16 v = (_Float16)0.0f;
        if (k < 20)       v = (_Float16)W1[k * 64 + n];
        else if (k == 20) v = (_Float16)b1[n];   // bias row, pairs with feat col 20 == 1
        w1t[n * 32 + k] = v;
    }
    for (int idx = tid; idx < 64 * 64; idx += 256) {
        int n = idx >> 6, k = idx & 63;
        w2t[n * 64 + k] = (_Float16)W2[k * 64 + n];
        w3t[n * 64 + k] = (_Float16)W3[k * 64 + n];
    }
    if (tid < 64) { sb2[tid] = b2[tid]; sb3[tid] = b3[tid]; }
    __syncthreads();

    const int wave  = tid >> 5;
    const int lane  = tid & 31;
    const int ln    = lane & 15;
    const int khalf = (lane < 16) ? 0 : 8;
    const int mrow0 = (lane < 16) ? 0 : 8;
    const int jj    = lane & 7;
    const int iAdd  = ln >> 3;

    _Float16* feat = featT[wave];
    _Float16* hB   = hT[wave];

    // loop-invariant per-lane bias values for layers 2 and 3
    float bv2[4], bv3[4];
#pragma unroll
    for (int j = 0; j < 4; j++) { bv2[j] = sb2[16 * j + ln]; bv3[j] = sb3[16 * j + ln]; }

    const long long totalTasks = (long long)B * tilesPerB;
    const long long gridWaves  = (long long)gridDim.x * 8;
    long long task = (long long)blockIdx.x * 8 + wave;

    for (; task < totalTasks; task += gridWaves) {
        const int b    = (int)(task / tilesPerB);
        const int tile = (int)(task % tilesPerB);

        // ---- build feature tile, split across all 32 lanes ----
        // Lanes 0..15: edge e -> cols 0..11  (p_src, p_dst, bx)
        // Lanes 16..31: same edge -> cols 12..19 (by), col 20 = 1.0 (bias),
        //               cols 21..31 zero.
        {
            int e = tile * 16 + ln;
            bool valid = (e < E);
            _Float16* frow = feat + ln * 32;
            if (lane < 16) {
                if (valid) {
                    int d = neigh[e];        // dst
                    int s = neigh[E + e];    // src
                    dstT[wave][lane] = d;
                    if (b == 0) atomicAdd(&deg[d], 1.0f);
                    const float* ps = points + 2 * (size_t)s;
                    const float* pd = points + 2 * (size_t)d;
                    const float* bx = batch + ((size_t)b * N + s) * FIN;
                    v8h lo; v4h mid;
                    lo[0] = (_Float16)ps[0]; lo[1] = (_Float16)ps[1];
                    lo[2] = (_Float16)pd[0]; lo[3] = (_Float16)pd[1];
#pragma unroll
                    for (int i = 0; i < 4; i++) lo[4 + i] = (_Float16)bx[i];
#pragma unroll
                    for (int i = 0; i < 4; i++) mid[i] = (_Float16)bx[4 + i];
                    *((v8h*)frow)       = lo;   // cols 0..7
                    *((v4h*)(frow + 8)) = mid;  // cols 8..11
                } else {
                    dstT[wave][lane] = -1;
                    v8h z8 = {}; v4h z4 = {};
                    *((v8h*)frow) = z8;
                    *((v4h*)(frow + 8)) = z4;
                }
            } else {
                v4h m12 = {}; v8h m16 = {}; v8h z8 = {};
                if (valid) {
                    int d = neigh[e];
                    const float* by = batch + ((size_t)b * N + d) * FIN;
#pragma unroll
                    for (int i = 0; i < 4; i++) m12[i] = (_Float16)by[i];
#pragma unroll
                    for (int i = 0; i < 4; i++) m16[i] = (_Float16)by[4 + i];
                    m16[4] = (_Float16)1.0f;   // col 20: bias multiplier
                }
                *((v4h*)(frow + 12)) = m12;  // cols 12..15
                *((v8h*)(frow + 16)) = m16;  // cols 16..23 (by[4..7], 1, 0..)
                *((v8h*)(frow + 24)) = z8;   // cols 24..31
            }
        }
        asm volatile("s_wait_dscnt 0" ::: "memory");

        v8f acc[4];

        // ---- layer 1: (16x32) x (32x64), 4 WMMAs, C = inline 0 ----
        {
            v16h a0 = ldsFrag(feat + ln * 32, 0, khalf);
#pragma unroll
            for (int j = 0; j < 4; j++) {
                v8f c = {};
                v16h bf = ldsFrag(w1t + (16 * j + ln) * 32, 0, khalf);
                acc[j] = wmma_f16(a0, bf, c);
            }
        }
#pragma unroll
        for (int j = 0; j < 4; j++)
#pragma unroll
            for (int r = 0; r < 8; r++)
                hB[(mrow0 + r) * 64 + 16 * j + ln] = (_Float16)fast_tanh(acc[j][r]);
        asm volatile("s_wait_dscnt 0" ::: "memory");

        // ---- layer 2: (16x64) x (64x64), 8 WMMAs ----
        {
            v16h a1 = ldsFrag(hB + ln * 64, 0,  khalf);
            v16h a2 = ldsFrag(hB + ln * 64, 32, khalf);
#pragma unroll
            for (int j = 0; j < 4; j++) {
                v8f c;
#pragma unroll
                for (int r = 0; r < 8; r++) c[r] = bv2[j];
                v16h bf0 = ldsFrag(w2t + (16 * j + ln) * 64, 0,  khalf);
                v16h bf1 = ldsFrag(w2t + (16 * j + ln) * 64, 32, khalf);
                c = wmma_f16(a1, bf0, c);
                c = wmma_f16(a2, bf1, c);
                acc[j] = c;
            }
        }
#pragma unroll
        for (int j = 0; j < 4; j++)
#pragma unroll
            for (int r = 0; r < 8; r++)
                hB[(mrow0 + r) * 64 + 16 * j + ln] = (_Float16)fast_tanh(acc[j][r]);
        asm volatile("s_wait_dscnt 0" ::: "memory");

        // ---- layer 3 (no tanh): mats in C layout, 8 WMMAs ----
        {
            v16h a1 = ldsFrag(hB + ln * 64, 0,  khalf);
            v16h a2 = ldsFrag(hB + ln * 64, 32, khalf);
#pragma unroll
            for (int j = 0; j < 4; j++) {
                v8f c;
#pragma unroll
                for (int r = 0; r < 8; r++) c[r] = bv3[j];
                v16h bf0 = ldsFrag(w3t + (16 * j + ln) * 64, 0,  khalf);
                v16h bf1 = ldsFrag(w3t + (16 * j + ln) * 64, 32, khalf);
                c = wmma_f16(a1, bf0, c);
                c = wmma_f16(a2, bf1, c);
                acc[j] = c;
            }
        }

        // ---- msg[m][i] = sum_j mats[m][i][j] * by[m][j]; DPP8 butterfly
        // reduce over the 8-lane j groups, then f32 atomic scatter-add.
        // C layout: tile t, lane L, vgpr r -> col n = 16t + (L&15), edge
        // m = r + (L<16?0:8). i = 2t + ((L&15)>>3); j = L&7.
        float byv[8];
#pragma unroll
        for (int r = 0; r < 8; r++)
            byv[r] = (float)feat[(mrow0 + r) * 32 + 12 + jj];

#pragma unroll
        for (int t = 0; t < 4; t++) {
            float p[8];
#pragma unroll
            for (int r = 0; r < 8; r++) p[r] = red8(acc[t][r] * byv[r]);
            if (jj == 0) {
                int i = 2 * t + iAdd;
#pragma unroll
                for (int r = 0; r < 8; r++) {
                    int d = dstT[wave][mrow0 + r];
                    if (d >= 0)
                        atomicAdd(&ksum[((size_t)b * N + d) * FOUT + i], p[r]);
                }
            }
        }
    }
}

// out[b][n][:] = batch[b][n][:] @ Wlin + ksum[b][n][:] / (deg[n] + 1)
__global__ void __launch_bounds__(256)
finalize_kernel(const float* __restrict__ batch, const float* __restrict__ Wlin,
                const float* __restrict__ ksum, const float* __restrict__ deg,
                float* __restrict__ out, int B, int N)
{
    __shared__ float wl[64];
    if (threadIdx.x < 64) wl[threadIdx.x] = Wlin[threadIdx.x];
    __syncthreads();

    long long t = (long long)blockIdx.x * blockDim.x + threadIdx.x;
    long long total = (long long)B * N;
    if (t >= total) return;
    int n = (int)(t % N);
    const float* br = batch + t * FIN;
    float x[FIN];
#pragma unroll
    for (int fi = 0; fi < FIN; fi++) x[fi] = br[fi];
    float inv = 1.0f / (deg[n] + 1.0f);
    const float* kr = ksum + t * FOUT;
#pragma unroll
    for (int fo = 0; fo < FOUT; fo++) {
        float o = kr[fo] * inv;
#pragma unroll
        for (int fi = 0; fi < FIN; fi++) o += x[fi] * wl[fi * FOUT + fo];
        out[t * FOUT + fo] = o;
    }
}

extern "C" void kernel_launch(void* const* d_in, const int* in_sizes, int n_in,
                              void* d_out, int out_size, void* d_ws, size_t ws_size,
                              hipStream_t stream)
{
    (void)n_in; (void)out_size; (void)ws_size;
    const float* batch  = (const float*)d_in[0];
    const float* points = (const float*)d_in[1];
    const int*   neigh  = (const int*)d_in[2];
    const float* W1 = (const float*)d_in[3];
    const float* b1 = (const float*)d_in[4];
    const float* W2 = (const float*)d_in[5];
    const float* b2 = (const float*)d_in[6];
    const float* W3 = (const float*)d_in[7];
    const float* b3 = (const float*)d_in[8];
    const float* Wlin = (const float*)d_in[9];
    float* out = (float*)d_out;

    const int E = in_sizes[2] / 2;           // neigh (2,E)
    const int N = in_sizes[1] / 2;           // points (N,2)
    const int B = in_sizes[0] / (N * FIN);   // batch (B,N,F_IN)

    float* ksum = (float*)d_ws;                       // B*N*FOUT floats
    float* deg  = ksum + (size_t)B * N * FOUT;        // N floats (contiguous)

    long long nz = (long long)B * N * FOUT + N;
    zero_kernel<<<(int)((nz + 255) / 256), 256, 0, stream>>>(ksum, nz);

    int tiles = (E + 15) / 16;
    long long tasks = (long long)B * tiles;
    long long wavesNeeded = (tasks + TASKS_PER_WAVE - 1) / TASKS_PER_WAVE;
    int blocks = (int)((wavesNeeded + 7) / 8);
    if (blocks < 1) blocks = 1;
    msg_kernel<<<blocks, 256, 0, stream>>>(batch, points, neigh,
                                           W1, b1, W2, b2, W3, b3,
                                           ksum, deg, B, N, E, tiles);

    long long tot = (long long)B * N;
    finalize_kernel<<<(int)((tot + 255) / 256), 256, 0, stream>>>(
        batch, Wlin, ksum, deg, out, B, N);
}
